// PVConv_16681652977956
// MI455X (gfx1250) — matrix operand order repaired and checked by reading, not compiled, not run on hardware
//
#include <hip/hip_runtime.h>
#include <hip/hip_bf16.h>
#include <stdint.h>

// ---------------- problem constants ----------------
#define B_    8
#define C_    64          // CIN == COUT == 64
#define N_    32768
#define R_    32
#define R3_   32768
#define WPAD_ 34          // padded innermost spatial dim (zero halo at 0 and 33)

// ---------------- workspace layout (bytes) ----------------
#define OFF_ACCUM   ((size_t)0)                         // [B][R3][64] f32 scatter accum; reused as conv2-out [B][64][R3]
#define SZ_ACCUM    ((size_t)B_*R3_*C_*4)               // 67,108,864
#define OFF_CNT     (OFF_ACCUM + SZ_ACCUM)              // [B][R3] f32 counts
#define SZ_CNT      ((size_t)B_*R3_*4)                  // 1,048,576
#define OFF_GRIDP   (OFF_CNT + SZ_CNT)                  // [B][32][32][34][64] bf16 padded grid; reused as featT [B][N][64] bf16
#define SZ_GRIDP    ((size_t)B_*R_*R_*WPAD_*C_*2)       // 35,651,584
#define OFF_CONV1   (OFF_GRIDP + SZ_GRIDP)              // [B][64][R3] f32 conv1 out; reused as point-branch p
#define SZ_CONV1    ((size_t)B_*C_*R3_*4)               // 67,108,864
#define OFF_NORMF   (OFF_CONV1 + SZ_CONV1)              // [B][N][3] f32 normalized coords
#define SZ_NORMF    ((size_t)B_*N_*3*4)
#define OFF_WP1     (OFF_NORMF + SZ_NORMF)              // packed conv1 weights: 4*27*2*32*8 dwords
#define SZ_WP1      ((size_t)4*27*2*32*8*4)             // 221,184
#define OFF_WP2     (OFF_WP1 + SZ_WP1)
#define SZ_WP2      SZ_WP1
#define OFF_WPP     (OFF_WP2 + SZ_WP2)                  // packed point weights: 4*1*2*32*8 dwords
#define SZ_WPP      ((size_t)4*1*2*32*8*4)              // 8,192
#define OFF_STATS   (OFF_WPP + SZ_WPP)                  // small stats region
#define SZ_STATS    ((size_t)4096)

// stats region float indices
#define ST_MEAN 0      // [B][3] coord means
#define ST_MAG  24     // [B] max magnitudes
#define ST_BN1  32     // sum[64] then sumsq[64]
#define ST_BN2  160
#define ST_BNP  288

typedef __attribute__((ext_vector_type(16))) __bf16   v16bf;
typedef __attribute__((ext_vector_type(8)))  float    v8f;
typedef __attribute__((ext_vector_type(4)))  unsigned uv4;

union Frag { uv4 q[2]; v16bf v; };

__device__ __forceinline__ unsigned short f2bf(float f) {
  union { float f; unsigned u; } v; v.f = f;
  unsigned u = v.u;
  u += 0x7fffu + ((u >> 16) & 1u);          // round-to-nearest-even
  return (unsigned short)(u >> 16);
}

// ---------------- coordinate statistics ----------------
__global__ void k_coord_mean(const float* __restrict__ coords, float* __restrict__ stats) {
  __shared__ float sh[256];
  int d = blockIdx.x, b = blockIdx.y;
  const float* p = coords + ((size_t)b * 3 + d) * N_;
  float s = 0.f;
  for (int i = threadIdx.x; i < N_; i += 256) s += p[i];
  sh[threadIdx.x] = s; __syncthreads();
  for (int off = 128; off > 0; off >>= 1) {
    if (threadIdx.x < off) sh[threadIdx.x] += sh[threadIdx.x + off];
    __syncthreads();
  }
  if (threadIdx.x == 0) stats[ST_MEAN + b * 3 + d] = sh[0] * (1.0f / N_);
}

__global__ void k_coord_mag(const float* __restrict__ coords, float* __restrict__ stats) {
  __shared__ float sh[256];
  int b = blockIdx.x;
  float m0 = stats[ST_MEAN + b*3 + 0];
  float m1 = stats[ST_MEAN + b*3 + 1];
  float m2 = stats[ST_MEAN + b*3 + 2];
  float mx = 0.f;
  for (int i = threadIdx.x; i < N_; i += 256) {
    float x = coords[((size_t)b*3 + 0) * N_ + i] - m0;
    float y = coords[((size_t)b*3 + 1) * N_ + i] - m1;
    float z = coords[((size_t)b*3 + 2) * N_ + i] - m2;
    mx = fmaxf(mx, sqrtf(x*x + y*y + z*z));
  }
  sh[threadIdx.x] = mx; __syncthreads();
  for (int off = 128; off > 0; off >>= 1) {
    if (threadIdx.x < off) sh[threadIdx.x] = fmaxf(sh[threadIdx.x], sh[threadIdx.x + off]);
    __syncthreads();
  }
  if (threadIdx.x == 0) stats[ST_MAG + b] = sh[0];
}

// ---------------- normalize coords + scatter-add voxelize ----------------
__global__ void k_norm_vox(const float* __restrict__ coords, const float* __restrict__ feats,
                           const float* __restrict__ stats, float* __restrict__ normf,
                           float* __restrict__ accum, float* __restrict__ cnt) {
  int tid = blockIdx.x * 256 + threadIdx.x;
  if (tid >= B_ * N_) return;
  int b = tid / N_, n = tid % N_;
  float inv = 1.0f / (stats[ST_MAG + b] * 2.0f);
  float nf[3]; int vi[3];
  #pragma unroll
  for (int d = 0; d < 3; ++d) {
    float c = coords[((size_t)b*3 + d) * N_ + n] - stats[ST_MEAN + b*3 + d];
    float v = (c * inv + 0.5f) * (float)R_;
    v = fminf(fmaxf(v, 0.0f), (float)(R_ - 1));
    nf[d] = v;
    vi[d] = (int)rintf(v);                       // round-half-even, matches jnp.round
  }
  normf[(size_t)tid*3 + 0] = nf[0];
  normf[(size_t)tid*3 + 1] = nf[1];
  normf[(size_t)tid*3 + 2] = nf[2];
  int flat = (vi[0] * R_ + vi[1]) * R_ + vi[2];
  atomicAdd(&cnt[(size_t)b * R3_ + flat], 1.0f);
  float* acc = accum + ((size_t)b * R3_ + flat) * C_;
  const float* f = feats + (size_t)b * C_ * N_ + n;
  #pragma unroll 8
  for (int c = 0; c < C_; ++c) atomicAdd(&acc[c], f[(size_t)c * N_]);
}

// divide by counts, convert to bf16 channel-last padded grid
__global__ void k_vox_finalize(const float* __restrict__ accum, const float* __restrict__ cnt,
                               unsigned short* __restrict__ gridp) {
  int tid = blockIdx.x * 256 + threadIdx.x;
  if (tid >= B_ * R3_) return;
  int b = tid >> 15, flat = tid & (R3_ - 1);
  int d0 = flat >> 10, d1 = (flat >> 5) & 31, d2 = flat & 31;
  float inv = 1.0f / fmaxf(cnt[tid], 1.0f);
  const float* a = accum + (size_t)tid * C_;
  unsigned short* g = gridp + (((size_t)((b*R_ + d0)*R_ + d1)) * WPAD_ + (d2 + 1)) * C_;
  #pragma unroll 8
  for (int c = 0; c < C_; ++c) g[c] = f2bf(a[c] * inv);
}

// ---------------- weight repack into WMMA A-fragment layout ----------------
// dst dword tid layout: frag = (mt*ntaps + t)*2 + s ; per frag: 32 lanes x 8 dwords (lane-contiguous)
// A 16x32 bf16, contiguous k order used here (same as B): lane<16 holds K=0..15 ; lane>=16 holds K=16..31
__global__ void k_pack_a(const float* __restrict__ w, unsigned* __restrict__ dst, int ntaps) {
  int tid = blockIdx.x * 256 + threadIdx.x;
  int total = 4 * ntaps * 2 * 32 * 8;
  if (tid >= total) return;
  int v    = tid & 7;
  int lane = (tid >> 3) & 31;
  int s    = (tid >> 8) & 1;
  int rest = tid >> 9;
  int t  = rest % ntaps;
  int mt = rest / ntaps;
  int m  = lane & 15;
  int kb = s * 32 + ((lane >= 16) ? 16 : 0) + v * 2;
  int co = mt * 16 + m;
  float f0 = w[((size_t)co * C_ + kb    ) * ntaps + t];
  float f1 = w[((size_t)co * C_ + kb + 1) * ntaps + t];
  dst[tid] = (unsigned)f2bf(f0) | ((unsigned)f2bf(f1) << 16);
}

// ---------------- conv3d 3x3x3, 64->64, implicit GEMM via WMMA bf16 ----------------
// one wave per (b, z, y, wtile of 16, co-tile of 16): 54 x v_wmma_f32_16x16x32_bf16
__global__ void __launch_bounds__(256) k_conv_wmma(const unsigned short* __restrict__ gridp,
                                                   const unsigned* __restrict__ apack,
                                                   const float* __restrict__ bias,
                                                   float* __restrict__ out) {
  int wave = (blockIdx.x * 256 + threadIdx.x) >> 5;
  int lane = threadIdx.x & 31;
  int mt = wave & 3;
  int wt = (wave >> 2) & 1;
  int y  = (wave >> 3) & 31;
  int z  = (wave >> 8) & 31;
  int b  = wave >> 13;
  int n  = lane & 15;        // N column == w position in tile
  int kh = lane >> 4;        // which K half this lane holds
  int w  = wt * 16 + n;
  v8f acc = {};
  #pragma unroll
  for (int t = 0; t < 27; ++t) {
    int dz = t / 9 - 1, dy = (t / 3) % 3 - 1, dx = t % 3 - 1;
    int zz = z + dz, yy = y + dy;
    if ((unsigned)zz >= 32u || (unsigned)yy >= 32u) continue;   // wave-uniform skip
    const unsigned short* vox =
        gridp + (((size_t)((b*R_ + zz)*R_ + yy)) * WPAD_ + (w + dx + 1)) * C_ + kh * 16;
    const unsigned* ap = apack + ((mt * 27 + t) * 2) * 256 + lane * 8;
    #pragma unroll
    for (int s = 0; s < 2; ++s) {
      Frag A, Bf;
      const uv4* pb = (const uv4*)(vox + s * 32);
      Bf.q[0] = pb[0]; Bf.q[1] = pb[1];
      const uv4* pa = (const uv4*)(ap + s * 256);
      A.q[0] = pa[0]; A.q[1] = pa[1];
      acc = __builtin_amdgcn_wmma_f32_16x16x32_bf16(false, A.v, false, Bf.v,
                                                    (short)0, acc, false, false);
    }
  }
  int spatial = (z * 32 + y) * 32 + w;
  size_t base = ((size_t)b * C_) << 15;
  #pragma unroll
  for (int r = 0; r < 8; ++r) {
    int co = mt * 16 + kh * 8 + r;   // C/D layout: lanes 0-15 M=r, lanes 16-31 M=8+r
    out[base + ((size_t)co << 15) + spatial] = acc[r] + bias[co];
  }
}

// ---------------- per-channel sum / sumsq over [B][64][32768] ----------------
__global__ void k_channel_stats(const float* __restrict__ x, float* __restrict__ stats, int sbase) {
  __shared__ float ss[256], sq[256];
  int c = blockIdx.x, b = blockIdx.y;
  const float* p = x + (((size_t)b * C_ + c) << 15);
  float s = 0.f, q = 0.f;
  for (int i = threadIdx.x; i < R3_; i += 256) { float v = p[i]; s += v; q += v * v; }
  ss[threadIdx.x] = s; sq[threadIdx.x] = q; __syncthreads();
  for (int off = 128; off > 0; off >>= 1) {
    if (threadIdx.x < off) { ss[threadIdx.x] += ss[threadIdx.x + off]; sq[threadIdx.x] += sq[threadIdx.x + off]; }
    __syncthreads();
  }
  if (threadIdx.x == 0) { atomicAdd(&stats[sbase + c], ss[0]); atomicAdd(&stats[sbase + 64 + c], sq[0]); }
}

// BN + leaky, f32 [B][64][R3] -> bf16 channel-last padded grid (for the next conv)
__global__ void k_bn_act_pad(const float* __restrict__ x, const float* __restrict__ stats, int sbase,
                             const float* __restrict__ g, const float* __restrict__ beta,
                             unsigned short* __restrict__ gridp) {
  int tid = blockIdx.x * 256 + threadIdx.x;
  if (tid >= B_ * R3_) return;
  int b = tid >> 15, flat = tid & (R3_ - 1);
  int d0 = flat >> 10, d1 = (flat >> 5) & 31, d2 = flat & 31;
  unsigned short* gp = gridp + (((size_t)((b*R_ + d0)*R_ + d1)) * WPAD_ + d2 + 1) * C_;
  const float invM = 1.0f / ((float)B_ * (float)R3_);
  for (int c = 0; c < C_; ++c) {
    float mean = stats[sbase + c] * invM;
    float var  = stats[sbase + 64 + c] * invM - mean * mean;
    float rs   = rsqrtf(var + 1e-4f);
    float sc   = g[c] * rs;
    float sh   = beta[c] - mean * sc;
    float v = x[(((size_t)b * C_ + c) << 15) + flat] * sc + sh;
    gp[c] = f2bf(v >= 0.f ? v : 0.1f * v);
  }
}

// BN + leaky in place, f32 [B][64][R3]
__global__ void k_bn_act_f32(float* __restrict__ x, const float* __restrict__ stats, int sbase,
                             const float* __restrict__ g, const float* __restrict__ beta) {
  size_t tid = (size_t)blockIdx.x * 256 + threadIdx.x;
  if (tid >= (size_t)B_ * C_ * R3_) return;
  int c = (int)((tid >> 15) & 63);
  const float invM = 1.0f / ((float)B_ * (float)R3_);
  float mean = stats[sbase + c] * invM;
  float var  = stats[sbase + 64 + c] * invM - mean * mean;
  float rs   = rsqrtf(var + 1e-4f);
  float sc   = g[c] * rs;
  float sh   = beta[c] - mean * sc;
  float v = x[tid] * sc + sh;
  x[tid] = v >= 0.f ? v : 0.1f * v;
}

// ---------------- trilinear devoxelize ----------------
__global__ void k_devox(const float* __restrict__ grid2, const float* __restrict__ normf,
                        float* __restrict__ out) {
  int tid = blockIdx.x * 256 + threadIdx.x;
  if (tid >= B_ * N_) return;
  int b = tid / N_, n = tid % N_;
  float nx = normf[(size_t)tid*3 + 0];
  float ny = normf[(size_t)tid*3 + 1];
  float nz = normf[(size_t)tid*3 + 2];
  int lx = (int)floorf(nx), ly = (int)floorf(ny), lz = (int)floorf(nz);
  float fx = nx - lx, fy = ny - ly, fz = nz - lz;
  int hx = min(lx + 1, R_ - 1), hy = min(ly + 1, R_ - 1), hz = min(lz + 1, R_ - 1);
  int   idx[8];
  float wg[8];
  #pragma unroll
  for (int k = 0; k < 8; ++k) {
    int dx = (k >> 2) & 1, dy = (k >> 1) & 1, dz = k & 1;
    int X = dx ? hx : lx, Y = dy ? hy : ly, Z = dz ? hz : lz;
    idx[k] = (X * R_ + Y) * R_ + Z;
    wg[k]  = (dx ? fx : 1.f - fx) * (dy ? fy : 1.f - fy) * (dz ? fz : 1.f - fz);
  }
  const float* gbase = grid2 + (((size_t)b * C_) << 15);
  float* obase = out + (size_t)b * C_ * N_ + n;
  for (int c = 0; c < C_; ++c) {
    const float* gc = gbase + ((size_t)c << 15);
    float s = 0.f;
    #pragma unroll
    for (int k = 0; k < 8; ++k) s += gc[idx[k]] * wg[k];
    obase[(size_t)c * N_] = s;
  }
}

// ---------------- point branch: transpose features to [B][N][64] bf16 ----------------
__global__ void k_feat_t(const float* __restrict__ f, unsigned short* __restrict__ ft) {
  int tid = blockIdx.x * 256 + threadIdx.x;
  if (tid >= B_ * N_) return;
  int b = tid / N_, n = tid % N_;
  unsigned short* o = ft + (size_t)tid * C_;
  const float* p = f + (size_t)b * C_ * N_ + n;
  #pragma unroll 8
  for (int c = 0; c < C_; ++c) o[c] = f2bf(p[(size_t)c * N_]);
}

// 1x1 point GEMM via WMMA: p[b,co,n] = sum_ci wp[co,ci]*feat[b,ci,n] + bp[co]
__global__ void __launch_bounds__(256) k_point_gemm(const unsigned short* __restrict__ ft,
                                                    const unsigned* __restrict__ apack,
                                                    const float* __restrict__ bias,
                                                    float* __restrict__ out) {
  int wave = (blockIdx.x * 256 + threadIdx.x) >> 5;
  int lane = threadIdx.x & 31;
  int mt = wave & 3;
  int nt = (wave >> 2) & 2047;
  int b  = wave >> 13;
  int n = lane & 15, kh = lane >> 4;
  int pt = nt * 16 + n;
  const unsigned short* col = ft + ((size_t)b * N_ + pt) * C_ + kh * 16;
  v8f acc = {};
  #pragma unroll
  for (int s = 0; s < 2; ++s) {
    Frag A, Bf;
    const uv4* pb = (const uv4*)(col + s * 32);
    Bf.q[0] = pb[0]; Bf.q[1] = pb[1];
    const uv4* pa = (const uv4*)(apack + (mt * 2 + s) * 256 + lane * 8);
    A.q[0] = pa[0]; A.q[1] = pa[1];
    acc = __builtin_amdgcn_wmma_f32_16x16x32_bf16(false, A.v, false, Bf.v,
                                                  (short)0, acc, false, false);
  }
  #pragma unroll
  for (int r = 0; r < 8; ++r) {
    int co = mt * 16 + kh * 8 + r;
    out[((size_t)b * C_ + co) * N_ + pt] = acc[r] + bias[co];
  }
}

// BN + ReLU on point branch, accumulate into output
__global__ void k_point_bn_add(const float* __restrict__ p, const float* __restrict__ stats,
                               const float* __restrict__ g, const float* __restrict__ beta,
                               float* __restrict__ out) {
  size_t tid = (size_t)blockIdx.x * 256 + threadIdx.x;
  if (tid >= (size_t)B_ * C_ * N_) return;
  int c = (int)((tid >> 15) & 63);
  const float invM = 1.0f / ((float)B_ * (float)N_);
  float mean = stats[ST_BNP + c] * invM;
  float var  = stats[ST_BNP + 64 + c] * invM - mean * mean;
  float rs   = rsqrtf(var + 1e-4f);
  float sc   = g[c] * rs;
  float sh   = beta[c] - mean * sc;
  float v = p[tid] * sc + sh;
  out[tid] += (v > 0.f ? v : 0.f);
}

// ---------------- launcher ----------------
extern "C" void kernel_launch(void* const* d_in, const int* in_sizes, int n_in,
                              void* d_out, int out_size, void* d_ws, size_t ws_size,
                              hipStream_t stream) {
  (void)in_sizes; (void)n_in; (void)out_size; (void)ws_size;
  const float* feats  = (const float*)d_in[0];
  const float* coords = (const float*)d_in[1];
  const float* w1  = (const float*)d_in[2];
  const float* b1  = (const float*)d_in[3];
  const float* g1  = (const float*)d_in[4];
  const float* be1 = (const float*)d_in[5];
  const float* w2  = (const float*)d_in[6];
  const float* b2  = (const float*)d_in[7];
  const float* g2  = (const float*)d_in[8];
  const float* be2 = (const float*)d_in[9];
  const float* wp  = (const float*)d_in[10];
  const float* bp  = (const float*)d_in[11];
  const float* gp  = (const float*)d_in[12];
  const float* bep = (const float*)d_in[13];
  float* out = (float*)d_out;

  char* ws = (char*)d_ws;
  float*          accum  = (float*)(ws + OFF_ACCUM);     // later: conv2 out
  float*          cnt    = (float*)(ws + OFF_CNT);
  unsigned short* gridp  = (unsigned short*)(ws + OFF_GRIDP);  // later: featT
  float*          conv1  = (float*)(ws + OFF_CONV1);     // later: point-branch p
  float*          normf  = (float*)(ws + OFF_NORMF);
  unsigned*       apack1 = (unsigned*)(ws + OFF_WP1);
  unsigned*       apack2 = (unsigned*)(ws + OFF_WP2);
  unsigned*       apackp = (unsigned*)(ws + OFF_WPP);
  float*          stats  = (float*)(ws + OFF_STATS);

  // zero scatter accumulators, padded-grid halo, stats (capture-safe memset nodes)
  hipMemsetAsync(accum, 0, SZ_ACCUM + SZ_CNT, stream);
  hipMemsetAsync(gridp, 0, SZ_GRIDP, stream);
  hipMemsetAsync(stats, 0, SZ_STATS, stream);

  // weight repack
  k_pack_a<<<216, 256, 0, stream>>>(w1, apack1, 27);
  k_pack_a<<<216, 256, 0, stream>>>(w2, apack2, 27);
  k_pack_a<<<8,   256, 0, stream>>>(wp, apackp, 1);

  // coordinate normalization + voxelization
  k_coord_mean<<<dim3(3, B_), 256, 0, stream>>>(coords, stats);
  k_coord_mag <<<B_,          256, 0, stream>>>(coords, stats);
  k_norm_vox  <<<(B_*N_)/256, 256, 0, stream>>>(coords, feats, stats, normf, accum, cnt);
  k_vox_finalize<<<(B_*R3_)/256, 256, 0, stream>>>(accum, cnt, gridp);

  // conv1 + BN + leaky -> bf16 padded grid
  k_conv_wmma<<<(B_*R_*R_*2*4)/8, 256, 0, stream>>>(gridp, apack1, b1, conv1);
  k_channel_stats<<<dim3(64, B_), 256, 0, stream>>>(conv1, stats, ST_BN1);
  k_bn_act_pad<<<(B_*R3_)/256, 256, 0, stream>>>(conv1, stats, ST_BN1, g1, be1, gridp);

  // conv2 + BN + leaky (f32, in place in accum buffer)
  k_conv_wmma<<<(B_*R_*R_*2*4)/8, 256, 0, stream>>>(gridp, apack2, b2, accum);
  k_channel_stats<<<dim3(64, B_), 256, 0, stream>>>(accum, stats, ST_BN2);
  k_bn_act_f32<<<(B_*C_*R3_)/256, 256, 0, stream>>>(accum, stats, ST_BN2, g2, be2);

  // devoxelize into output
  k_devox<<<(B_*N_)/256, 256, 0, stream>>>(accum, normf, out);

  // point branch: transpose -> WMMA GEMM -> BN + ReLU -> add
  k_feat_t<<<(B_*N_)/256, 256, 0, stream>>>(feats, gridp);
  k_point_gemm<<<(B_*2048*4)/8, 256, 0, stream>>>(gridp, apackp, bp, conv1);
  k_channel_stats<<<dim3(64, B_), 256, 0, stream>>>(conv1, stats, ST_BNP);
  k_point_bn_add<<<(B_*C_*N_)/256, 256, 0, stream>>>(conv1, stats, gp, bep, out);
}
